// GATv2FeatureExtractor_37555194036837
// MI455X (gfx1250) — matrix operand order repaired and checked by reading, not compiled, not run on hardware
//
#include <hip/hip_runtime.h>
#include <hip/hip_bf16.h>
#include <math.h>

// ---------------------------------------------------------------------------
// CDNA5 (gfx1250) GATv2 feature extractor.
//  - Node GEMMs: v_wmma_f32_16x16x32_bf16, templated on (K, Nc) so k-loops
//    fully unroll and all fragment loads are 128-bit vector loads.
//  - Weights pre-packed to bf16 in the exact per-lane B-fragment layout.
//  - Edge phase (bandwidth-bound): fused edge-attr GEMM, CAS float-max,
//    exp + atomic-add segment softmax, float4 gather + atomic scatter.
// ---------------------------------------------------------------------------

typedef __attribute__((ext_vector_type(16))) __bf16 v16bf;
typedef __attribute__((ext_vector_type(8)))  float  v8f;

#define NEG_SLOPE 0.2f

// ---------------- float atomic max via CAS ----------------
__device__ __forceinline__ void atomicMaxFloat(float* addr, float val) {
  int* ai = (int*)addr;
  int old = __float_as_int(*addr);
  while (__int_as_float(old) < val) {
    int assumed = old;
    old = atomicCAS(ai, assumed, __float_as_int(val));
    if (old == assumed) break;
  }
}

// ---------------- weight pre-pack: fp32 [K,Nc] -> bf16 B-fragment layout ----
// Dword index t encodes: d (0..7) | lane (0..31) | nTile | kTile.
// Element pair in dword d of lane L, tile (kT,nT):
//   kb = (L>=16)*8 ; k = kT*32 + kb + 2(d&3) + (d>>2)*16 (+1) ; n = nT*16 + (L&15)
__global__ void pack_b_kernel(const float* __restrict__ B, unsigned int* __restrict__ P,
                              int K, int Nc) {
  int total = (K >> 5) * (Nc >> 4) * 256;
  int t = blockIdx.x * blockDim.x + threadIdx.x;
  if (t >= total) return;
  int d = t & 7;
  int lane = (t >> 3) & 31;
  int nTiles = Nc >> 4;
  int nT = (t >> 8) % nTiles;
  int kT = (t >> 8) / nTiles;
  int kb = ((lane >> 4) & 1) * 8;
  int k = kT * 32 + kb + 2 * (d & 3) + (d >> 2) * 16;
  int n = nT * 16 + (lane & 15);
  union { __bf16 b[2]; unsigned int u; } p;
  p.b[0] = (__bf16)B[(size_t)k * Nc + n];
  p.b[1] = (__bf16)B[(size_t)(k + 1) * Nc + n];
  P[t] = p.u;
}

// ---------------- GEMM: C[M,Nc] = act(A[M,K] @ B[K,Nc] + bias) -------------
// A fp32 row-major; B pre-packed bf16 fragments; one wave per 16x16 C tile.
template <int K, int Nc>
__global__ __launch_bounds__(256)
void gemm_wmma(const float* __restrict__ A, const unsigned int* __restrict__ Bpk,
               const float* __restrict__ bias, float* __restrict__ C,
               int M, int act) {
  constexpr int nTiles = Nc >> 4;
  constexpr int kTiles = K >> 5;
  int wave = blockIdx.x * (blockDim.x >> 5) + (threadIdx.x >> 5);
  int mTiles = M >> 4;
  if (wave >= mTiles * nTiles) return;
  int mTile = wave / nTiles;
  int nTile = wave % nTiles;
  int lane = threadIdx.x & 31;

  // A layout: lanes 0-15 -> M=lane, K {0..7,16..23}; lanes 16-31 -> K {8..15,24..31}
  int kbA = ((lane >> 4) & 1) * 8;
  size_t rowBase = (size_t)(mTile * 16 + (lane & 15)) * (size_t)K;
  const uint4* bbase = (const uint4*)Bpk;

  v8f acc = {};
#pragma unroll
  for (int kT = 0; kT < kTiles; ++kT) {
    // ---- A fragment: 4 x float4 contiguous loads, convert to bf16 ----
    const float4* ap = (const float4*)(A + rowBase + kT * 32 + kbA);
    float4 f0 = ap[0];   // k = kb+0..3
    float4 f1 = ap[1];   // k = kb+4..7
    float4 f2 = ap[4];   // k = kb+16..19
    float4 f3 = ap[5];   // k = kb+20..23
    v16bf a;
    a[0]  = (__bf16)f0.x; a[1]  = (__bf16)f0.y; a[2]  = (__bf16)f0.z; a[3]  = (__bf16)f0.w;
    a[4]  = (__bf16)f1.x; a[5]  = (__bf16)f1.y; a[6]  = (__bf16)f1.z; a[7]  = (__bf16)f1.w;
    a[8]  = (__bf16)f2.x; a[9]  = (__bf16)f2.y; a[10] = (__bf16)f2.z; a[11] = (__bf16)f2.w;
    a[12] = (__bf16)f3.x; a[13] = (__bf16)f3.y; a[14] = (__bf16)f3.z; a[15] = (__bf16)f3.w;

    // ---- B fragment: 2 x uint4 loads of pre-packed bf16 ----
    size_t bidx = ((size_t)(kT * nTiles + nTile) * 32 + lane) * 2;
    union { uint4 q[2]; v16bf v; } ub;
    ub.q[0] = bbase[bidx];
    ub.q[1] = bbase[bidx + 1];

    acc = __builtin_amdgcn_wmma_f32_16x16x32_bf16(
        false, a, false, ub.v, (short)0, acc, false, false);
  }

  // C layout: VGPR r -> row mTile*16 + (lane<16 ? r : r+8), col = nTile*16 + (lane&15)
  int n = nTile * 16 + (lane & 15);
  float bv = bias ? bias[n] : 0.0f;
  int rbase = mTile * 16 + ((lane < 16) ? 0 : 8);
#pragma unroll
  for (int r = 0; r < 8; ++r) {
    float v = acc[r] + bv;
    if (act) v = fmaxf(v, 0.0f);
    C[(size_t)(rbase + r) * (size_t)Nc + n] = v;
  }
}

// ---------------- fill ----------------
__global__ void fill_kernel(float* __restrict__ p, float v, long long n) {
  long long i = (long long)blockIdx.x * blockDim.x + threadIdx.x;
  long long stride = (long long)gridDim.x * blockDim.x;
  for (; i < n; i += stride) p[i] = v;
}

// ---------------- attention logits (edge GEMM fused) + segment max ---------
template <int H, int C>
__global__ void edge_alpha_kernel(const int* __restrict__ src, const int* __restrict__ dst,
                                  const float* __restrict__ ea,   // [E,8]
                                  const float* __restrict__ We,   // [8, H*C]
                                  const float* __restrict__ xl,   // [N, H*C]
                                  const float* __restrict__ xr,   // [N, H*C]
                                  const float* __restrict__ att,  // [H*C]
                                  float* __restrict__ alpha,      // [E*H]
                                  float* __restrict__ amax,       // [N*H]
                                  int E) {
  constexpr int HC = H * C;
  long long t = (long long)blockIdx.x * blockDim.x + threadIdx.x;
  if (t >= (long long)E * H) return;
  int e = (int)(t / H);
  int h = (int)(t % H);
  int s = src[e];
  int d = dst[e];

  float ev[8];
  {
    const float4* ea4 = (const float4*)(ea + (size_t)e * 8);
    float4 e0 = ea4[0], e1 = ea4[1];
    ev[0] = e0.x; ev[1] = e0.y; ev[2] = e0.z; ev[3] = e0.w;
    ev[4] = e1.x; ev[5] = e1.y; ev[6] = e1.z; ev[7] = e1.w;
  }

  const float4* xl4 = (const float4*)(xl + (size_t)s * HC + (size_t)h * C);
  const float4* xr4 = (const float4*)(xr + (size_t)d * HC + (size_t)h * C);
  const float4* at4 = (const float4*)(att + (size_t)h * C);
  const float4* we4 = (const float4*)(We + (size_t)h * C);
  constexpr int WSTRIDE = HC / 4;   // float4 stride between We rows

  float acc = 0.0f;
  for (int c4 = 0; c4 < C / 4; ++c4) {
    float4 xlv = xl4[c4];
    float4 xrv = xr4[c4];
    float4 atv = at4[c4];
    float4 et = make_float4(0.f, 0.f, 0.f, 0.f);
#pragma unroll
    for (int j = 0; j < 8; ++j) {
      float4 wv = we4[(size_t)j * WSTRIDE + c4];
      et.x = fmaf(ev[j], wv.x, et.x);
      et.y = fmaf(ev[j], wv.y, et.y);
      et.z = fmaf(ev[j], wv.z, et.z);
      et.w = fmaf(ev[j], wv.w, et.w);
    }
    float v0 = xlv.x + xrv.x + et.x;
    float v1 = xlv.y + xrv.y + et.y;
    float v2 = xlv.z + xrv.z + et.z;
    float v3 = xlv.w + xrv.w + et.w;
    v0 = (v0 > 0.f) ? v0 : NEG_SLOPE * v0;
    v1 = (v1 > 0.f) ? v1 : NEG_SLOPE * v1;
    v2 = (v2 > 0.f) ? v2 : NEG_SLOPE * v2;
    v3 = (v3 > 0.f) ? v3 : NEG_SLOPE * v3;
    acc = fmaf(v0, atv.x, acc);
    acc = fmaf(v1, atv.y, acc);
    acc = fmaf(v2, atv.z, acc);
    acc = fmaf(v3, atv.w, acc);
  }
  alpha[t] = acc;
  atomicMaxFloat(&amax[(size_t)d * H + h], acc);
}

// ---------------- exp + segment sum ----------------
template <int H>
__global__ void edge_exp_kernel(const int* __restrict__ dst, const float* __restrict__ amax,
                                float* __restrict__ alpha, float* __restrict__ denom,
                                int E) {
  long long t = (long long)blockIdx.x * blockDim.x + threadIdx.x;
  if (t >= (long long)E * H) return;
  int e = (int)(t / H);
  int h = (int)(t % H);
  int d = dst[e];
  float ex = __expf(alpha[t] - amax[(size_t)d * H + h]);
  alpha[t] = ex;
  atomicAdd(&denom[(size_t)d * H + h], ex);
}

// ---------------- weighted scatter-aggregate (4 channels / thread) ---------
template <int H, int C>
__global__ void edge_aggr_kernel(const int* __restrict__ src, const int* __restrict__ dst,
                                 const float* __restrict__ xl,
                                 const float* __restrict__ alpha,
                                 const float* __restrict__ denom,
                                 float* __restrict__ acc, int E) {
  constexpr int HC = H * C;
  constexpr int G = HC >> 2;
  long long t = (long long)blockIdx.x * blockDim.x + threadIdx.x;
  if (t >= (long long)E * G) return;
  int e = (int)(t / G);
  int j = (int)(t % G);
  int c0 = j << 2;
  int h = c0 / C;
  int s = src[e];
  int d = dst[e];
  float a = alpha[(size_t)e * H + h] / denom[(size_t)d * H + h];
  const float4 xv = *(const float4*)(xl + (size_t)s * HC + c0);
  float* op = acc + (size_t)d * HC + c0;
  atomicAdd(op + 0, xv.x * a);
  atomicAdd(op + 1, xv.y * a);
  atomicAdd(op + 2, xv.z * a);
  atomicAdd(op + 3, xv.w * a);
}

// ---------------- bias + relu in place (HC fixed at 256) ----------------
__global__ void finalize_kernel(float* __restrict__ p, const float* __restrict__ bias,
                                long long n) {
  long long i = (long long)blockIdx.x * blockDim.x + threadIdx.x;
  long long stride = (long long)gridDim.x * blockDim.x;
  for (; i < n; i += stride) {
    float v = p[i] + bias[(int)(i & 255)];
    p[i] = fmaxf(v, 0.0f);
  }
}

// ---------------------------------------------------------------------------
extern "C" void kernel_launch(void* const* d_in, const int* in_sizes, int n_in,
                              void* d_out, int out_size, void* d_ws, size_t ws_size,
                              hipStream_t stream) {
  const float* x    = (const float*)d_in[0];
  const int*   ei   = (const int*)d_in[1];
  const float* ea   = (const float*)d_in[2];
  const float* W1   = (const float*)d_in[3];
  const float* b1   = (const float*)d_in[4];
  const float* W2   = (const float*)d_in[5];
  const float* b2   = (const float*)d_in[6];
  const float* Wl1  = (const float*)d_in[7];
  const float* bl1  = (const float*)d_in[8];
  const float* Wr1  = (const float*)d_in[9];
  const float* br1  = (const float*)d_in[10];
  const float* We1  = (const float*)d_in[11];
  const float* att1 = (const float*)d_in[12];
  const float* bias1= (const float*)d_in[13];
  const float* Wl2  = (const float*)d_in[14];
  const float* bl2  = (const float*)d_in[15];
  const float* Wr2  = (const float*)d_in[16];
  const float* br2  = (const float*)d_in[17];
  const float* We2  = (const float*)d_in[18];
  const float* att2 = (const float*)d_in[19];
  const float* bias2= (const float*)d_in[20];
  (void)n_in; (void)out_size; (void)ws_size;

  const int N = in_sizes[0] / 32;   // 50000
  const int E = in_sizes[2] / 8;    // 800000
  const int* src = ei;
  const int* dstv = ei + E;

  // ---------------- workspace layout (floats / dwords) ----------------
  float* ws = (float*)d_ws;
  size_t off = 0;
  float* t_buf   = ws + off; off += (size_t)N * 64;   // MLP stage 1
  float* h_buf   = ws + off; off += (size_t)N * 64;   // MLP output
  float* xl_buf  = ws + off; off += (size_t)N * 256;  // source-side transform
  float* xr_buf  = ws + off; off += (size_t)N * 256;  // dest-side transform
  float* acc_buf = ws + off; off += (size_t)N * 256;  // layer-1 aggregate / h1
  float* alpha   = ws + off; off += (size_t)E * 4;    // logits / exp
  float* amax    = ws + off; off += (size_t)N * 4;    // segment max
  float* denom   = ws + off; off += (size_t)N * 4;    // segment sum
  // packed bf16 weights (dword counts = K*Nc/2)
  unsigned int* pkW1  = (unsigned int*)(ws + off); off += 32 * 64 / 2;
  unsigned int* pkW2  = (unsigned int*)(ws + off); off += 64 * 64 / 2;
  unsigned int* pkWl1 = (unsigned int*)(ws + off); off += 64 * 256 / 2;
  unsigned int* pkWr1 = (unsigned int*)(ws + off); off += 64 * 256 / 2;
  unsigned int* pkWl2 = (unsigned int*)(ws + off); off += 256 * 256 / 2;
  unsigned int* pkWr2 = (unsigned int*)(ws + off); off += 256 * 256 / 2;
  float* outp = (float*)d_out;

  auto fillf = [&](float* p, float v, long long n) {
    long long b = (n + 255) / 256;
    int blocks = (int)(b > 16384 ? 16384 : b);
    fill_kernel<<<blocks, 256, 0, stream>>>(p, v, n);
  };
  auto packb = [&](const float* B, unsigned int* P, int K, int Nc) {
    int total = (K >> 5) * (Nc >> 4) * 256;
    pack_b_kernel<<<(total + 255) / 256, 256, 0, stream>>>(B, P, K, Nc);
  };

  // ---------------- pack all weights to bf16 fragments ----------------
  packb(W1, pkW1, 32, 64);
  packb(W2, pkW2, 64, 64);
  packb(Wl1, pkWl1, 64, 256);
  packb(Wr1, pkWr1, 64, 256);
  packb(Wl2, pkWl2, 256, 256);
  packb(Wr2, pkWr2, 256, 256);

  const int mTiles = N >> 4;   // N divisible by 16 (50000 = 3125*16)

  // ---------------- MLP encoder ----------------
  gemm_wmma<32, 64><<<(mTiles * 4 + 7) / 8, 256, 0, stream>>>(x, pkW1, b1, t_buf, N, 1);
  gemm_wmma<64, 64><<<(mTiles * 4 + 7) / 8, 256, 0, stream>>>(t_buf, pkW2, b2, h_buf, N, 1);

  // ---------------- GAT layer 1 (H=4, C=64) ----------------
  fillf(amax, -INFINITY, (long long)N * 4);
  fillf(denom, 0.0f,     (long long)N * 4);
  fillf(acc_buf, 0.0f,   (long long)N * 256);
  gemm_wmma<64, 256><<<(mTiles * 16 + 7) / 8, 256, 0, stream>>>(h_buf, pkWl1, bl1, xl_buf, N, 0);
  gemm_wmma<64, 256><<<(mTiles * 16 + 7) / 8, 256, 0, stream>>>(h_buf, pkWr1, br1, xr_buf, N, 0);
  {
    long long T = (long long)E * 4;
    int blocks = (int)((T + 255) / 256);
    edge_alpha_kernel<4, 64><<<blocks, 256, 0, stream>>>(src, dstv, ea, We1, xl_buf, xr_buf,
                                                         att1, alpha, amax, E);
    edge_exp_kernel<4><<<blocks, 256, 0, stream>>>(dstv, amax, alpha, denom, E);
  }
  {
    long long T = (long long)E * 64;   // E * (HC/4)
    int blocks = (int)((T + 255) / 256);
    edge_aggr_kernel<4, 64><<<blocks, 256, 0, stream>>>(src, dstv, xl_buf, alpha, denom,
                                                        acc_buf, E);
  }
  {
    long long n = (long long)N * 256;
    long long b = (n + 255) / 256;
    int blocks = (int)(b > 16384 ? 16384 : b);
    finalize_kernel<<<blocks, 256, 0, stream>>>(acc_buf, bias1, n);
  }

  // ---------------- GAT layer 2 (H=1, C=256) ----------------
  fillf(amax, -INFINITY, (long long)N);
  fillf(denom, 0.0f,     (long long)N);
  fillf(outp, 0.0f,      (long long)N * 256);
  gemm_wmma<256, 256><<<(mTiles * 16 + 7) / 8, 256, 0, stream>>>(acc_buf, pkWl2, bl2, xl_buf, N, 0);
  gemm_wmma<256, 256><<<(mTiles * 16 + 7) / 8, 256, 0, stream>>>(acc_buf, pkWr2, br2, xr_buf, N, 0);
  {
    long long T = (long long)E;
    int blocks = (int)((T + 255) / 256);
    edge_alpha_kernel<1, 256><<<blocks, 256, 0, stream>>>(src, dstv, ea, We2, xl_buf, xr_buf,
                                                          att2, alpha, amax, E);
    edge_exp_kernel<1><<<blocks, 256, 0, stream>>>(dstv, amax, alpha, denom, E);
  }
  {
    long long T = (long long)E * 64;
    int blocks = (int)((T + 255) / 256);
    edge_aggr_kernel<1, 256><<<blocks, 256, 0, stream>>>(src, dstv, xl_buf, alpha, denom,
                                                         outp, E);
  }
  {
    long long n = (long long)N * 256;
    long long b = (n + 255) / 256;
    int blocks = (int)(b > 16384 ? 16384 : b);
    finalize_kernel<<<blocks, 256, 0, stream>>>(outp, bias2, n);
  }
}
